// CACISLoss_78761110274122
// MI455X (gfx1250) — matrix operand order repaired and checked by reading, not compiled, not run on hardware
//
#include <hip/hip_runtime.h>

// CACIS loss for B=512, K=256 on gfx1250.
// One workgroup (256 threads = 8 wave32) per batch row. C[b] is read from HBM
// exactly once; the K x K matrix lives in LDS (fp16, 128 KB of the 320 KB WGP
// LDS). Frank-Wolfe matvecs use v_wmma_f32_16x16x32_f16 with the alpha vector
// broadcast across the 16 B-matrix columns.

typedef __attribute__((ext_vector_type(16))) _Float16 v16h;
typedef __attribute__((ext_vector_type(8)))  _Float16 v8h;
typedef __attribute__((ext_vector_type(8)))  float    v8f;

#define K      256
#define BATCH  512
#define NITER  50

__device__ __forceinline__ unsigned orderable(float v) {
    unsigned u = __float_as_uint(v);
    return (u & 0x80000000u) ? ~u : (u | 0x80000000u);
}

__device__ __forceinline__ float unorderable(unsigned r) {
    unsigned u = (r & 0x80000000u) ? (r ^ 0x80000000u) : ~r;
    return __uint_as_float(u);
}

__global__ __launch_bounds__(256)
void cacis_row_kernel(const float* __restrict__ scores,
                      const int*   __restrict__ targets,
                      const float* __restrict__ C,
                      float*       __restrict__ losses) {
    extern __shared__ char smem[];
    unsigned long long* slot64   = (unsigned long long*)(smem + 0);
    float*              sumSlot  = (float*)(smem + 8);
    float*              trSlot   = (float*)(smem + 12);
    unsigned*           minSlot  = (unsigned*)(smem + 16);
    float*              quadSlot = (float*)(smem + 20);
    float*              sc       = (float*)(smem + 32);            // 256 f32
    float*              alpha    = (float*)(smem + 32 + 1024);     // 256 f32
    _Float16*           alphah   = (_Float16*)(smem + 2080);       // 256 f16
    float*              gvec     = (float*)(smem + 2592);          // 256 f32
    _Float16*           Mh       = (_Float16*)(smem + 4096);       // 256*256 f16

    const int tid  = threadIdx.x;
    const int b    = blockIdx.x;
    const int lane = tid & 31;
    const int wave = tid >> 5;
    const int row_lo = lane & 15;
    const int hi16   = (lane >> 4) & 1;

    if (tid == 0) { *sumSlot = 0.f; *trSlot = 0.f; *minSlot = 0xFFFFFFFFu; }
    sc[tid] = 0.5f * scores[b * K + tid];
    __syncthreads();

    // ---- Pass 1: single HBM sweep of C[b]. Thread t owns column j = t. ----
    const float* Cb = C + (size_t)b * K * K;
    const float fj = sc[tid];
    float psum = 0.f, pdiag = 0.f, pmin = 3.4e38f;
    for (int r = 0; r < K; ++r) {
        float c   = Cb[r * K + tid];
        float fij = sc[r] + fj + c;
        psum += c;
        if (r == tid) pdiag = c;
        pmin = fminf(pmin, fij);
        Mh[r * K + tid] = (_Float16)fij;          // cache pre-exp value in LDS
    }
    atomicAdd(sumSlot, psum);
    atomicAdd(trSlot, pdiag);
    atomicMin(minSlot, orderable(pmin));
    __syncthreads();

    const float sum  = *sumSlot;
    const float tr   = *trSlot;
    const float minv = unorderable(*minSlot);
    const float eps  = fmaxf((sum - tr) / (float)(K * K - K), 1e-8f);
    const float shift  = -minv / eps;
    const float inveps = 1.0f / eps;

    // ---- Pass 2 (LDS only): M = exp(-fij/eps - shift), in (0,1]. ----
    for (int r = 0; r < K; ++r) {
        float fij = (float)Mh[r * K + tid];
        Mh[r * K + tid] = (_Float16)__expf(-fij * inveps - shift);
    }
    alpha[tid]  = 1.0f / K;
    alphah[tid] = (_Float16)(1.0f / K);
    __syncthreads();

    // ---- Frank-Wolfe: 50 iterations + 1 final matvec for alpha^T M alpha ----
    for (int it = 0; it <= NITER; ++it) {
        if (tid == 0) *slot64 = 0xFFFFFFFFFFFFFFFFull;
        __syncthreads();

        float gmin = 3.4e38f;
        int   gidx = 0;
        for (int rt = 0; rt < 2; ++rt) {
            const int i0  = (wave * 2 + rt) * 16;
            const int row = i0 + row_lo;
            v8f acc = {};
            #pragma unroll
            for (int kt = 0; kt < 8; ++kt) {
                const int k0 = kt * 32;
                // A operand: 16x32 fp16 tile of M (documented A layout).
                const _Float16* rp = &Mh[row * K + k0 + (hi16 ? 8 : 0)];
                v8h alo = *(const v8h*)(rp);
                v8h ahi = *(const v8h*)(rp + 16);
                v16h a;
                #pragma unroll
                for (int h = 0; h < 8; ++h) { a[h] = alo[h]; a[h + 8] = ahi[h]; }
                // B operand: alpha broadcast to all 16 columns (32x16 layout).
                const _Float16* bp = &alphah[k0 + (hi16 ? 8 : 0)];
                v8h blo = *(const v8h*)(bp);
                v8h bhi = *(const v8h*)(bp + 16);
                v16h bb;
                #pragma unroll
                for (int h = 0; h < 8; ++h) { bb[h] = blo[h]; bb[h + 8] = bhi[h]; }
                acc = __builtin_amdgcn_wmma_f32_16x16x32_f16(
                        false, a, false, bb, (short)0, acc, false, false);
            }
            if (it < NITER) {
                #pragma unroll
                for (int r = 0; r < 8; ++r) {
                    const int rowi = i0 + r + (hi16 ? 8 : 0);
                    const float g = acc[r];          // = (M*alpha)[rowi]
                    if (g < gmin) { gmin = g; gidx = rowi; }
                }
            } else if (row_lo == 0) {
                #pragma unroll
                for (int r = 0; r < 8; ++r) gvec[i0 + r + (hi16 ? 8 : 0)] = acc[r];
            }
        }

        if (it < NITER) {
            // workgroup argmin (first-index tie-break) via packed u64 LDS min
            unsigned long long key =
                ((unsigned long long)orderable(gmin) << 32) | (unsigned)gidx;
            atomicMin(slot64, key);
            __syncthreads();
            const int idx = (int)(unsigned)(*slot64 & 0xFFFFFFFFull);
            const float gamma = 2.0f / ((float)it + 2.0f);
            float a2 = alpha[tid] * (1.0f - gamma);
            if (tid == idx) a2 += gamma;
            alpha[tid]  = a2;
            alphah[tid] = (_Float16)a2;
            __syncthreads();
        }
    }
    __syncthreads();

    // quad = alpha^T M alpha  (zero-alpha entries contribute 0, matching the
    // reference's -inf masked logsumexp)
    if (tid == 0) *quadSlot = 0.f;
    __syncthreads();
    atomicAdd(quadSlot, alpha[tid] * gvec[tid]);
    __syncthreads();

    if (tid == 0) {
        const float quad    = *quadSlot;
        const float logval  = __logf(quad);
        const float conj    = -eps * (logval + shift);
        const float fy      = scores[b * K + targets[b]];
        losses[b] = conj - fy;
    }
}

__global__ __launch_bounds__(256)
void cacis_reduce_kernel(const float* __restrict__ losses,
                         float* __restrict__ out) {
    __shared__ float red[256];
    const int t = threadIdx.x;
    red[t] = losses[t] + losses[t + 256];
    __syncthreads();
    #pragma unroll
    for (int s = 128; s > 0; s >>= 1) {
        if (t < s) red[t] += red[t + s];
        __syncthreads();
    }
    if (t == 0) out[0] = red[0] * (1.0f / (float)BATCH);
}

extern "C" void kernel_launch(void* const* d_in, const int* in_sizes, int n_in,
                              void* d_out, int out_size, void* d_ws, size_t ws_size,
                              hipStream_t stream) {
    (void)in_sizes; (void)n_in; (void)out_size; (void)ws_size;
    const float* scores  = (const float*)d_in[0];
    const int*   targets = (const int*)d_in[1];
    const float* C       = (const float*)d_in[2];
    float*       losses  = (float*)d_ws;

    const size_t shmem = 4096 + (size_t)K * K * sizeof(_Float16); // ~132 KB
    // Best-effort opt-in for >64KB dynamic LDS (gfx1250 WGP has 320 KB).
    (void)hipFuncSetAttribute((const void*)cacis_row_kernel,
                              hipFuncAttributeMaxDynamicSharedMemorySize,
                              (int)shmem);

    cacis_row_kernel<<<BATCH, 256, shmem, stream>>>(scores, targets, C, losses);
    cacis_reduce_kernel<<<1, 256, 0, stream>>>(losses, (float*)d_out);
}
